// RNN_90056874263172
// MI455X (gfx1250) — compile-verified, matrix operand order checked by reading, not run on hardware
//
#include <hip/hip_runtime.h>

typedef _Float16 v16h __attribute__((ext_vector_type(16)));
typedef _Float16 v8h  __attribute__((ext_vector_type(8)));
typedef float    v8f  __attribute__((ext_vector_type(8)));
typedef float    v4f  __attribute__((ext_vector_type(4)));

static constexpr int kDIN  = 256;
static constexpr int kH    = 512;
static constexpr int kDOUT = 128;
static constexpr int kB    = 64;
static constexpr int kT    = 2048;
static constexpr int kLDH  = kH + 8;   // padded LDS row stride in halves (bank-conflict-free)

__device__ __forceinline__ v8f zero8() {
    v8f z;
#pragma unroll
    for (int i = 0; i < 8; ++i) z[i] = 0.0f;
    return z;
}

// tanh(x) = 1 - 2/(exp(2x)+1), via v_exp_f32 (base-2) and v_rcp_f32.
// Saturates correctly: x>>0 -> exp2=inf -> rcp=0 -> 1; x<<0 -> exp2=0 -> 1-2 = -1.
__device__ __forceinline__ float fast_tanhf(float x) {
    float e = __builtin_amdgcn_exp2f(x * 2.8853900817779268f);  // 2*log2(e)
    return 1.0f - 2.0f * __builtin_amdgcn_rcpf(e + 1.0f);
}

// Load one 16-half WMMA fragment row from f32 memory.
// Per ISA 7.12.2 (16-bit A/B 16x32): this lane needs elements
// [off..off+7] and [off+16..off+23] of its row.
__device__ __forceinline__ v16h load_frag_f32(const float* __restrict__ p) {
    v4f f0 = *(const v4f*)(p);
    v4f f1 = *(const v4f*)(p + 4);
    v4f f2 = *(const v4f*)(p + 16);
    v4f f3 = *(const v4f*)(p + 20);
    v16h r;
#pragma unroll
    for (int i = 0; i < 4; ++i) {
        r[i]      = (_Float16)f0[i];
        r[4 + i]  = (_Float16)f1[i];
        r[8 + i]  = (_Float16)f2[i];
        r[12 + i] = (_Float16)f3[i];
    }
    return r;
}

// Load an A fragment (16 halves) from LDS h-buffer: two b128 reads.
__device__ __forceinline__ v16h load_frag_lds(const _Float16* __restrict__ hb, int off) {
    v8h lo = *(const v8h*)(hb + off);
    v8h hi = *(const v8h*)(hb + off + 16);
    v16h r;
#pragma unroll
    for (int i = 0; i < 8; ++i) { r[i] = lo[i]; r[8 + i] = hi[i]; }
    return r;
}

// ---------------------------------------------------------------------------
// Kernel 1: x_proj[t, b, n] = sum_k X[b,t,k] * W_ih[n,k] + b_ih[n] + b_hh[n]
// Stored TIME-MAJOR [T, B, H] so the scan kernel touches one [B,H] slab/step.
// Grid: 8192 blocks x 256 threads. Block = 16 rows of (B*T); wave = 64 cols.
// ---------------------------------------------------------------------------
__global__ __launch_bounds__(256) void rnn_xproj_kernel(
    const float* __restrict__ X, const float* __restrict__ W_ih,
    const float* __restrict__ b_ih, const float* __restrict__ b_hh,
    float* __restrict__ xp) {
    const int lane  = threadIdx.x & 31;
    const int wave  = threadIdx.x >> 5;            // 0..7 -> 64-col group
    const int mtile = blockIdx.x;                  // 0..8191
    const int l15   = lane & 15;
    const int khalf = (lane >> 4) ? 8 : 0;         // hi-lane K offset
    const int arow  = mtile * 16 + l15;            // A (X) row = b*T + t
    const int n0    = wave * 64 + l15;

    v8f acc[4];
#pragma unroll
    for (int nt = 0; nt < 4; ++nt) acc[nt] = zero8();

#pragma unroll
    for (int kt = 0; kt < kDIN / 32; ++kt) {       // 8 K-tiles
        const int klo = kt * 32 + khalf;
        v16h a = load_frag_f32(X + (size_t)arow * kDIN + klo);
#pragma unroll
        for (int nt = 0; nt < 4; ++nt) {
            const int n = n0 + nt * 16;
            v16h b = load_frag_f32(W_ih + (size_t)n * kDIN + klo);
            acc[nt] = __builtin_amdgcn_wmma_f32_16x16x32_f16(
                false, a, false, b, (short)0, acc[nt], false, false);
        }
    }

    const int rbase = (lane >> 4) ? 8 : 0;         // C-fragment row offset
#pragma unroll
    for (int nt = 0; nt < 4; ++nt) {
        const int n = n0 + nt * 16;
        const float bias = b_ih[n] + b_hh[n];
#pragma unroll
        for (int r = 0; r < 8; ++r) {
            const int g  = mtile * 16 + r + rbase; // output row = b*T + t
            const int bb = g >> 11;                // / T
            const int tt = g & (kT - 1);           // % T
            xp[((size_t)tt * kB + bb) * kH + n] = acc[nt][r] + bias;
        }
    }
}

// ---------------------------------------------------------------------------
// Kernel 2: the sequential scan + final projection.
// Grid: 4 blocks x 1024 threads (32 waves). Block = 16 batch rows.
// Wave w owns output columns [16w, 16w+16); W_hh^T fragments (16 K-tiles,
// 128 VGPRs/lane) are preloaded into registers and reused for all 2048 steps.
// h state lives in LDS as f16, padded row stride for conflict-free b128 reads.
// xp is register double-buffered: step t+1's 8 strided loads are issued at
// the top of step t, so their L2/HBM latency hides behind the 16-WMMA chain
// and the barrier instead of sitting on the post-barrier critical path.
// ---------------------------------------------------------------------------
__global__ __launch_bounds__(1024) void rnn_scan_kernel(
    const float* __restrict__ xp,       // [T, B, H]
    const float* __restrict__ W_hh,     // [H, H]
    const float* __restrict__ W_out,    // [D_OUT, H]
    const float* __restrict__ b_out,    // [D_OUT]
    float* __restrict__ y) {            // [B, D_OUT]
    __shared__ _Float16 hbuf[16 * kLDH];

    const int lane  = threadIdx.x & 31;
    const int wave  = threadIdx.x >> 5;     // 0..31
    const int l15   = lane & 15;
    const int khalf = (lane >> 4) ? 8 : 0;
    const int rbase = (lane >> 4) ? 8 : 0;
    const int b0    = blockIdx.x * 16;
    const int ncol  = wave * 16 + l15;      // this lane's output column

    // Preload W_hh^T as 16 B-fragments (B[k,n] = W_hh[n,k]; row n contiguous).
    v16h bw[16];
#pragma unroll
    for (int kt = 0; kt < 16; ++kt)
        bw[kt] = load_frag_f32(W_hh + (size_t)ncol * kH + kt * 32 + khalf);

    // Per-lane base for its 8 xp elements: rows (b0 + rbase + r), col ncol.
    const float* xbase = xp + (size_t)(b0 + rbase) * kH + ncol;

    // h0 = 0
    for (int i = threadIdx.x; i < 16 * kLDH; i += 1024) hbuf[i] = (_Float16)0.0f;

    // Preload xp for t = 0 while waves are still filling/zeroing.
    float cur[8];
#pragma unroll
    for (int r = 0; r < 8; ++r) cur[r] = xbase[(size_t)r * kH];

    __syncthreads();

    for (int t = 0; t < kT; ++t) {
        // Issue next step's xp loads first; clamp keeps them in-bounds and
        // branchless at t = T-1 (value unused there).
        const int tn = (t + 1 < kT) ? (t + 1) : t;
        const float* xnext = xbase + (size_t)tn * kB * kH;
        float nxt[8];
#pragma unroll
        for (int r = 0; r < 8; ++r) nxt[r] = xnext[(size_t)r * kH];

        v8f acc0 = zero8(), acc1 = zero8();
#pragma unroll
        for (int kt = 0; kt < 16; kt += 2) {
            v16h a0 = load_frag_lds(hbuf, l15 * kLDH + kt * 32 + khalf);
            acc0 = __builtin_amdgcn_wmma_f32_16x16x32_f16(
                false, a0, false, bw[kt], (short)0, acc0, false, false);
            v16h a1 = load_frag_lds(hbuf, l15 * kLDH + (kt + 1) * 32 + khalf);
            acc1 = __builtin_amdgcn_wmma_f32_16x16x32_f16(
                false, a1, false, bw[kt + 1], (short)0, acc1, false, false);
        }
        __syncthreads();  // everyone done reading h_{t-1}

#pragma unroll
        for (int r = 0; r < 8; ++r) {
            const int m = r + rbase;
            float v = acc0[r] + acc1[r] + cur[r];
            hbuf[m * kLDH + ncol] = (_Float16)fast_tanhf(v);
        }
        __syncthreads();  // h_t visible to all waves

#pragma unroll
        for (int r = 0; r < 8; ++r) cur[r] = nxt[r];
    }

    // y = h_final @ W_out^T + b_out : 16 x 128 per block, waves 0..7.
    if (wave < 8) {
        v8f acc = zero8();
#pragma unroll
        for (int kt = 0; kt < 16; ++kt) {
            v16h a = load_frag_lds(hbuf, l15 * kLDH + kt * 32 + khalf);
            v16h b = load_frag_f32(W_out + (size_t)ncol * kH + kt * 32 + khalf);
            acc = __builtin_amdgcn_wmma_f32_16x16x32_f16(
                false, a, false, b, (short)0, acc, false, false);
        }
        const float bias = b_out[ncol];
#pragma unroll
        for (int r = 0; r < 8; ++r) {
            const int m = r + rbase;
            y[(size_t)(b0 + m) * kDOUT + ncol] = acc[r] + bias;
        }
    }
}

extern "C" void kernel_launch(void* const* d_in, const int* in_sizes, int n_in,
                              void* d_out, int out_size, void* d_ws, size_t ws_size,
                              hipStream_t stream) {
    (void)in_sizes; (void)n_in; (void)out_size; (void)ws_size;
    const float* X     = (const float*)d_in[0];
    const float* W_ih  = (const float*)d_in[1];
    const float* W_hh  = (const float*)d_in[2];
    const float* b_ih  = (const float*)d_in[3];
    const float* b_hh  = (const float*)d_in[4];
    const float* W_out = (const float*)d_in[5];
    const float* b_out = (const float*)d_in[6];
    float* xp = (float*)d_ws;                         // [T,B,H] f32 = 256 MB

    rnn_xproj_kernel<<<(kB * kT) / 16, 256, 0, stream>>>(X, W_ih, b_ih, b_hh, xp);
    rnn_scan_kernel<<<kB / 16, 1024, 0, stream>>>(xp, W_hh, W_out, b_out,
                                                  (float*)d_out);
}
